// XTAttention_56092272886438
// MI455X (gfx1250) — compile-verified, matrix operand order checked by reading, not run on hardware
//
#include <hip/hip_runtime.h>

typedef _Float16 f16;
typedef __attribute__((ext_vector_type(16))) _Float16 v16h;
typedef __attribute__((ext_vector_type(8)))  _Float16 v8h;
typedef __attribute__((ext_vector_type(8)))  float    v8f;

#define DIMN 1024
#define HEADD 64
#define NHEADS 16
#define BATCH 4
#define SEQ 2048
#define ROWS (BATCH*SEQ)   /* 8192 */
#define K3   (3*DIMN)      /* 3072 */
#define ATTN_SCALE 0.125f  /* 64^-0.5 */

// K-index mapping for 16-bit A/B fragments (ISA 7.12.2, 16-bit A 16x32):
// VGPR j holds 2 f16; lanes 0-15 cover K {0..7,16..23}, lanes 16-31 add +8.
__device__ __forceinline__ int pairK(int j) { return (j < 4) ? (2 * j) : (16 + 2 * (j - 4)); }

union AFrag { unsigned u[8]; v16h v; };

// CDNA5 async copy: global -> LDS, 16 bytes per lane, tracked by ASYNCcnt.
// LDS operand is the low 32 bits of the generic pointer (LDS aperture truncates
// addr[31:0] per ISA 10.2).
__device__ __forceinline__ void async_b128(unsigned lds_off, const void* g) {
  asm volatile("global_load_async_to_lds_b128 %0, %1, off"
               :: "v"(lds_off), "v"(g) : "memory");
}
__device__ __forceinline__ void wait_async0() {
  asm volatile("s_wait_asynccnt 0x0" ::: "memory");
}

// ---------------------------------------------------------------- pack kernels
__global__ void pack_xcat(const float* __restrict__ a, const float* __restrict__ b,
                          const float* __restrict__ c, f16* __restrict__ out) {
  size_t total = (size_t)ROWS * K3;
  for (size_t i = (size_t)blockIdx.x * blockDim.x + threadIdx.x; i < total;
       i += (size_t)gridDim.x * blockDim.x) {
    size_t row = i / K3;
    int col = (int)(i % K3);
    const float* src = (col < DIMN) ? a : (col < 2 * DIMN ? b : c);
    out[i] = (f16)src[row * DIMN + (col & (DIMN - 1))];
  }
}

__global__ void pack_w3(const float* __restrict__ a, const float* __restrict__ b,
                        const float* __restrict__ c, f16* __restrict__ out) {
  size_t blk = (size_t)DIMN * DIMN;
  size_t total = 3 * blk;
  for (size_t i = (size_t)blockIdx.x * blockDim.x + threadIdx.x; i < total;
       i += (size_t)gridDim.x * blockDim.x) {
    const float* s = (i < blk) ? a : (i < 2 * blk ? b : c);
    out[i] = (f16)s[i % blk];
  }
}

__global__ void pack_w1(const float* __restrict__ a, f16* __restrict__ out, size_t total) {
  for (size_t i = (size_t)blockIdx.x * blockDim.x + threadIdx.x; i < total;
       i += (size_t)gridDim.x * blockDim.x)
    out[i] = (f16)a[i];
}

// ---------------------------------------------------------------- WMMA GEMM
// C[M,N] = A[M,K] @ B[K,N]; f16 in, f32 accumulate; 128x128x64 tiles,
// 256 threads = 8 waves in 4(m) x 2(n) grid, each wave computes 32x64.
// A tile staged via global_load_async_to_lds_b128; B tile transposed via VGPRs.
template <bool OUT_F32, bool HAS_BIAS>
__global__ __launch_bounds__(256)
void gemm_wmma(const f16* __restrict__ A, const f16* __restrict__ B,
               void* __restrict__ Cout, const float* __restrict__ bias,
               int K, int lda, int ldb, int ldc) {
  __shared__ f16 lds_a[128 * 72];   // [row][k], padded stride (144B, 16B-aligned)
  __shared__ f16 lds_bt[128 * 72];  // B transposed: [n][k], padded stride

  const int tid = threadIdx.x;
  const int wave = tid >> 5, lane = tid & 31;
  const int m = lane & 15, half = lane >> 4;
  const int waveM = wave >> 1, waveN = wave & 1;
  const int rowBase = blockIdx.y * 128;
  const int colBase = blockIdx.x * 128;

  v8f acc[2][4] = {};

  for (int k0 = 0; k0 < K; k0 += 64) {
    __syncthreads();
    // A tile: 128x64 f16 = 1024 x 16B, async straight into LDS
#pragma unroll
    for (int it = 0; it < 4; ++it) {
      int v = tid + it * 256;
      int r = v >> 3;            // 8 vectors per row of 64
      int c = (v & 7) * 8;
      async_b128((unsigned)(size_t)&lds_a[r * 72 + c],
                 &A[(size_t)(rowBase + r) * lda + k0 + c]);
    }
    // B tile: 64x128 f16, transposed into LDS through VGPRs
#pragma unroll
    for (int it = 0; it < 4; ++it) {
      int v = tid + it * 256;
      int kr = v >> 4;           // 16 vectors per row of 128
      int c = (v & 15) * 8;
      v8h val = *(const v8h*)&B[(size_t)(k0 + kr) * ldb + colBase + c];
#pragma unroll
      for (int e = 0; e < 8; ++e) lds_bt[(c + e) * 72 + kr] = val[e];
    }
    wait_async0();
    __syncthreads();

#pragma unroll
    for (int ks = 0; ks < 2; ++ks) {
      const int kk = ks * 32;
      AFrag a[2], b[4];
#pragma unroll
      for (int ms = 0; ms < 2; ++ms) {
        int row = waveM * 32 + ms * 16 + m;
#pragma unroll
        for (int j = 0; j < 8; ++j)
          a[ms].u[j] = *(const unsigned*)&lds_a[row * 72 + kk + pairK(j) + 8 * half];
      }
#pragma unroll
      for (int ns = 0; ns < 4; ++ns) {
        int col = waveN * 64 + ns * 16 + m;
#pragma unroll
        for (int j = 0; j < 8; ++j)
          b[ns].u[j] = *(const unsigned*)&lds_bt[col * 72 + kk + pairK(j) + 8 * half];
      }
#pragma unroll
      for (int ms = 0; ms < 2; ++ms)
#pragma unroll
        for (int ns = 0; ns < 4; ++ns)
          acc[ms][ns] = __builtin_amdgcn_wmma_f32_16x16x32_f16(
              false, a[ms].v, false, b[ns].v, (short)0, acc[ms][ns], false, false);
    }
  }

  // Branch-free epilogue: C layout -> row = r + 8*half, col = lane&15.
#pragma unroll
  for (int ms = 0; ms < 2; ++ms)
#pragma unroll
    for (int ns = 0; ns < 4; ++ns) {
      int row0 = rowBase + waveM * 32 + ms * 16 + 8 * half;
      int col = colBase + waveN * 64 + ns * 16 + m;
      float bv = 0.f;
      if constexpr (HAS_BIAS) bv = bias[col];
#pragma unroll
      for (int r = 0; r < 8; ++r) {
        float val = acc[ms][ns][r] + bv;
        size_t idx = (size_t)(row0 + r) * ldc + col;
        if constexpr (OUT_F32) ((float*)Cout)[idx] = val;
        else                   ((f16*)Cout)[idx] = (f16)val;
      }
    }
}

// ---------------------------------------------------------------- flash attention
// grid = (SEQ/64, BATCH*NHEADS); 128 threads = 4 waves, each wave owns 16 q-rows.
__global__ __launch_bounds__(128)
void flash_attn(const f16* __restrict__ Q, const f16* __restrict__ Kmat,
                const f16* __restrict__ Vmat, f16* __restrict__ O) {
  __shared__ f16 lds_k[64 * 64];       // [key][d]
  __shared__ f16 lds_vt[64 * 64];      // [d][key]
  __shared__ f16 lds_p[4 * 16 * 64];   // per-wave P staging (C-layout -> A-layout)

  const int tid = threadIdx.x;
  const int wave = tid >> 5, lane = tid & 31;
  const int m = lane & 15, half = lane >> 4;
  const int bh = blockIdx.y, bi = bh >> 4, h = bh & 15;
  const int qrow0 = bi * SEQ + blockIdx.x * 64 + wave * 16;
  const int qcol0 = h * HEADD;

  // Q rows in A-fragment layout (two frags: d 0..31 and 32..63)
  AFrag aQ[2];
#pragma unroll
  for (int f = 0; f < 2; ++f)
#pragma unroll
    for (int j = 0; j < 8; ++j)
      aQ[f].u[j] = *(const unsigned*)
          &Q[(size_t)(qrow0 + m) * DIMN + qcol0 + pairK(j) + 8 * half + 32 * f];

  v8f Oacc[4] = {};
  float mrun[8], lrun[8];
#pragma unroll
  for (int r = 0; r < 8; ++r) { mrun[r] = -1e30f; lrun[r] = 0.f; }

  for (int kc = 0; kc < SEQ / 64; ++kc) {
    __syncthreads();
    // K chunk async -> LDS; V chunk transposed via VGPRs
#pragma unroll
    for (int it = 0; it < 4; ++it) {
      int v = tid + it * 128;
      int r = v >> 3;
      int c = (v & 7) * 8;
      size_t g = (size_t)(bi * SEQ + kc * 64 + r) * DIMN + qcol0 + c;
      async_b128((unsigned)(size_t)&lds_k[r * 64 + c], &Kmat[g]);
      v8h vv = *(const v8h*)&Vmat[g];
#pragma unroll
      for (int e = 0; e < 8; ++e) lds_vt[(c + e) * 64 + r] = vv[e];
    }
    wait_async0();
    __syncthreads();

    // S = Q @ K^T  (B frag: B[d][key], contiguous d in lds_k rows)
    v8f S[4];
#pragma unroll
    for (int s = 0; s < 4; ++s) {
      AFrag bK[2];
#pragma unroll
      for (int f = 0; f < 2; ++f)
#pragma unroll
        for (int j = 0; j < 8; ++j)
          bK[f].u[j] = *(const unsigned*)
              &lds_k[(s * 16 + m) * 64 + pairK(j) + 8 * half + 32 * f];
      v8f c = {};
      c = __builtin_amdgcn_wmma_f32_16x16x32_f16(false, aQ[0].v, false, bK[0].v,
                                                 (short)0, c, false, false);
      c = __builtin_amdgcn_wmma_f32_16x16x32_f16(false, aQ[1].v, false, bK[1].v,
                                                 (short)0, c, false, false);
      S[s] = c;
    }

    // online softmax; row r of the C tile lives in the 16 lanes of one half-wave,
    // so xor-shuffles 1/2/4/8 reduce across exactly those lanes.
#pragma unroll
    for (int r = 0; r < 8; ++r) {
      float mx = -1e30f;
#pragma unroll
      for (int s = 0; s < 4; ++s) { S[s][r] *= ATTN_SCALE; mx = fmaxf(mx, S[s][r]); }
#pragma unroll
      for (int off = 1; off <= 8; off <<= 1) mx = fmaxf(mx, __shfl_xor(mx, off, 32));
      float nm = fmaxf(mrun[r], mx);
      float alpha = __expf(mrun[r] - nm);
      mrun[r] = nm;
      float ssum = 0.f;
#pragma unroll
      for (int s = 0; s < 4; ++s) {
        float p = __expf(S[s][r] - nm);
        S[s][r] = p;
        ssum += p;
        Oacc[s][r] *= alpha;
      }
#pragma unroll
      for (int off = 1; off <= 8; off <<= 1) ssum += __shfl_xor(ssum, off, 32);
      lrun[r] = lrun[r] * alpha + ssum;
    }

    // P (C layout) -> LDS -> reload in A layout
#pragma unroll
    for (int s = 0; s < 4; ++s)
#pragma unroll
      for (int r = 0; r < 8; ++r)
        lds_p[wave * 1024 + (r + 8 * half) * 64 + s * 16 + m] = (f16)S[s][r];
    __syncthreads();

    AFrag aP[2];
#pragma unroll
    for (int f = 0; f < 2; ++f)
#pragma unroll
      for (int j = 0; j < 8; ++j)
        aP[f].u[j] = *(const unsigned*)
            &lds_p[wave * 1024 + m * 64 + pairK(j) + 8 * half + 32 * f];

    // O += P @ V  (B frag: B[key][d] from lds_vt, contiguous key)
#pragma unroll
    for (int s = 0; s < 4; ++s) {
      AFrag bV[2];
#pragma unroll
      for (int f = 0; f < 2; ++f)
#pragma unroll
        for (int j = 0; j < 8; ++j)
          bV[f].u[j] = *(const unsigned*)
              &lds_vt[(s * 16 + m) * 64 + pairK(j) + 8 * half + 32 * f];
      Oacc[s] = __builtin_amdgcn_wmma_f32_16x16x32_f16(false, aP[0].v, false, bV[0].v,
                                                       (short)0, Oacc[s], false, false);
      Oacc[s] = __builtin_amdgcn_wmma_f32_16x16x32_f16(false, aP[1].v, false, bV[1].v,
                                                       (short)0, Oacc[s], false, false);
    }
  }

  // normalize and store (f16, row-major [b*n, h*d])
#pragma unroll
  for (int s = 0; s < 4; ++s)
#pragma unroll
    for (int r = 0; r < 8; ++r) {
      float val = Oacc[s][r] / lrun[r];
      O[(size_t)(qrow0 + r + 8 * half) * DIMN + qcol0 + s * 16 + m] = (f16)val;
    }
}

// ---------------------------------------------------------------- launch
extern "C" void kernel_launch(void* const* d_in, const int* in_sizes, int n_in,
                              void* d_out, int out_size, void* d_ws, size_t ws_size,
                              hipStream_t stream) {
  (void)in_sizes; (void)n_in; (void)out_size; (void)ws_size;
  const float* words     = (const float*)d_in[0];
  const float* position  = (const float*)d_in[1];
  const float* conscious = (const float*)d_in[2];
  const float* w_wq = (const float*)d_in[3];
  const float* w_wk = (const float*)d_in[4];
  const float* w_wv = (const float*)d_in[5];
  const float* w_pq = (const float*)d_in[6];
  const float* w_pk = (const float*)d_in[7];
  const float* w_cq = (const float*)d_in[8];
  const float* w_ck = (const float*)d_in[9];
  const float* w_out = (const float*)d_in[10];
  const float* b_out = (const float*)d_in[11];

  f16* ws   = (f16*)d_ws;
  f16* Xcat = ws;                               // 8192 x 3072
  f16* Wqc  = Xcat + (size_t)ROWS * K3;         // 3072 x 1024 [w_wq; w_pq; w_cq]
  f16* Wkc  = Wqc + (size_t)K3 * DIMN;          // 3072 x 1024 [w_wk; w_pk; w_ck]
  f16* Wv   = Wkc + (size_t)K3 * DIMN;          // 1024 x 1024
  f16* Wo   = Wv + (size_t)DIMN * DIMN;         // 1024 x 1024
  f16* Qm   = Wo + (size_t)DIMN * DIMN;         // 8192 x 1024
  f16* Km   = Qm + (size_t)ROWS * DIMN;
  f16* Vm   = Km + (size_t)ROWS * DIMN;
  f16* Am   = Vm + (size_t)ROWS * DIMN;

  pack_xcat<<<4096, 256, 0, stream>>>(words, position, conscious, Xcat);
  pack_w3<<<1024, 256, 0, stream>>>(w_wq, w_pq, w_cq, Wqc);
  pack_w3<<<1024, 256, 0, stream>>>(w_wk, w_pk, w_ck, Wkc);
  pack_w1<<<512, 256, 0, stream>>>(w_wv, Wv, (size_t)DIMN * DIMN);
  pack_w1<<<512, 256, 0, stream>>>(w_out, Wo, (size_t)DIMN * DIMN);

  dim3 g(DIMN / 128, ROWS / 128);  // (8, 64)
  // Q = Xcat @ Wqc (K=3072); K = Xcat @ Wkc; V = words @ Wv (first 1024 cols)
  gemm_wmma<false, false><<<g, 256, 0, stream>>>(Xcat, Wqc, Qm, nullptr, K3, K3, DIMN, DIMN);
  gemm_wmma<false, false><<<g, 256, 0, stream>>>(Xcat, Wkc, Km, nullptr, K3, K3, DIMN, DIMN);
  gemm_wmma<false, false><<<g, 256, 0, stream>>>(Xcat, Wv, Vm, nullptr, DIMN, K3, DIMN, DIMN);

  flash_attn<<<dim3(SEQ / 64, BATCH * NHEADS), 128, 0, stream>>>(Qm, Km, Vm, Am);

  // out = Attn @ Wo + b_out  (f32 output)
  gemm_wmma<true, true><<<g, 256, 0, stream>>>(Am, Wo, d_out, b_out, DIMN, DIMN, DIMN, DIMN);
}
